// Beam_Search_76072460747150
// MI455X (gfx1250) — matrix operand order repaired and checked by reading, not compiled, not run on hardware
//
#include <hip/hip_runtime.h>
#include <hip/hip_bf16.h>
#include <math.h>
#include <cstddef>

// Problem constants (from reference)
#define V_     32000
#define E_     256
#define H_     512
#define L_     128
#define BB     16      // batch
#define BEAM_  2
#define MAXLEN 32
#define SOS_   2
#define EOS_   3
#define NEG_   (-1e30f)
#define G3     1536    // 3*H
#define EH     768     // E+H
#define F_     1280    // E+2H

typedef __attribute__((ext_vector_type(16))) _Float16 v16h;
typedef __attribute__((ext_vector_type(8)))  _Float16 v8h;
typedef __attribute__((ext_vector_type(8)))  float    v8f;

// ---------------- WMMA fragment helpers (wave32, 16x16x32 f16) ----------------
// A (16xK row-major): lane = hf*16+m holds row m; v[0..7]=A[m][kb+hf*8+j],
// v[8..15]=A[m][kb+16+hf*8+j]  (per ISA 16-bit A-matrix table)
// B (Kx16) = W^T tile: v[0..7]=W[n][kb+hf*8+j], v[8..15]=W[n][kb+16+hf*8+j] (same k order as A)

__device__ __forceinline__ v16h frag_from(const _Float16* p0, const _Float16* p1) {
  v8h lo = *(const v8h*)p0;
  v8h hi = *(const v8h*)p1;
  return __builtin_shufflevector(lo, hi, 0,1,2,3,4,5,6,7,8,9,10,11,12,13,14,15);
}
__device__ __forceinline__ v16h load_a(const _Float16* Abase, int ld, int m, int hf, int kb) {
  const _Float16* p = Abase + m * ld + kb + hf * 8;
  return frag_from(p, p + 16);
}
__device__ __forceinline__ v16h load_b(const _Float16* Wbase, int ld, int n, int hf, int kb) {
  const _Float16* p = Wbase + n * ld + kb + hf * 8;
  return frag_from(p, p + 16);
}
__device__ __forceinline__ v8f wmma16(v16h a, v16h b, v8f c) {
  // (neg_a, A, neg_b, B, c_mod, C, reuse_a, reuse_b) -> v_wmma_f32_16x16x32_f16
  return __builtin_amdgcn_wmma_f32_16x16x32_f16(false, a, false, b, (short)0, c, false, false);
}
// Fast nonlinearities: single v_exp_f32 + v_rcp_f32 (no IEEE divide chains on the
// latency-critical GRU recurrence).
__device__ __forceinline__ float sigmoidf_(float x) {
  return __builtin_amdgcn_rcpf(1.0f + __expf(-x));
}
__device__ __forceinline__ float tanhf_(float x) {
  // tanh(x) = 1 - 2/(exp(2x)+1); saturates to +/-1 correctly for large |x|
  return 1.0f - 2.0f * __builtin_amdgcn_rcpf(__expf(2.0f * x) + 1.0f);
}

// ---------------- weight conversion f32 -> f16 ----------------
__global__ void f32_to_f16_kernel(const float* __restrict__ src, _Float16* __restrict__ dst, long n) {
  long i = (long)blockIdx.x * blockDim.x + threadIdx.x;
  long stride = (long)gridDim.x * blockDim.x;
  for (; i < n; i += stride) dst[i] = (_Float16)src[i];
}

// ---------------- beam state init ----------------
__global__ void init_state_kernel(int* tok, float* prob, int* fin, int* th, float* sh) {
  int i = blockIdx.x * blockDim.x + threadIdx.x;
  if (i < BB * BEAM_) {
    tok[i]  = SOS_;
    prob[i] = (i & 1) ? NEG_ : 1.0f;   // beam0=1.0, beam1=NEG
    fin[i]  = 0;
  }
  if (i < BB * BEAM_ * MAXLEN) {
    int k = (i / MAXLEN) & 1;
    int j = i % MAXLEN;
    th[i] = (j == 0) ? SOS_ : EOS_;
    sh[i] = (k == 0 && j == 0) ? 1.0f : 0.0f;
  }
}

// ---------------- persistent encoder GRU: 1 block, 16 waves ----------------
// A-outer tiling: each wave loads one A fragment per k-step and drives 4 (rz)
// or 2+2 (n) independent WMMA accumulator chains. r,z gates fuse the Wih and
// Whh products into one accumulator (only i+h is needed for sigmoid gates).
__global__ __launch_bounds__(512) void enc_gru_kernel(
    const int* __restrict__ src, const float* __restrict__ emb_enc,
    const _Float16* __restrict__ Wih, const _Float16* __restrict__ Whh,
    const float* __restrict__ bih, const float* __restrict__ bhh,
    float* __restrict__ ctx_f32, _Float16* __restrict__ ctx_f16,
    float* __restrict__ hid_f32, _Float16* __restrict__ hid_f16)
{
  __shared__ __align__(16) _Float16 sX[BB][E_];   //  8 KB
  __shared__ __align__(16) _Float16 sH[BB][H_];   // 16 KB
  __shared__ __align__(16) _Float16 sR[BB][H_];   // 16 KB
  __shared__ __align__(16) _Float16 sZ[BB][H_];   // 16 KB
  __shared__ int stok[BB];

  const int tid  = threadIdx.x;
  const int wave = tid >> 5;
  const int lane = tid & 31;
  const int m  = lane & 15;
  const int hf = lane >> 4;

  // this wave's fixed gate columns (loop-invariant over the 128 steps)
  const int jrz = (wave << 6) + m;            // +q*16, q=0..3  (cols 0..1023)
  const int jn0 = 1024 + (wave << 5) + m;     // +q*16, q=0..1  (cols 1024..1535)
  float bsum_rz[4], bi_n[2], bh_n[2];
  #pragma unroll
  for (int q = 0; q < 4; ++q) bsum_rz[q] = bih[jrz + (q << 4)] + bhh[jrz + (q << 4)];
  #pragma unroll
  for (int q = 0; q < 2; ++q) { bi_n[q] = bih[jn0 + (q << 4)]; bh_n[q] = bhh[jn0 + (q << 4)]; }

  for (int i = tid; i < BB * H_; i += 512) (&sH[0][0])[i] = (_Float16)0.0f;
  __syncthreads();

  for (int t = 0; t < L_; ++t) {
    if (tid < BB) stok[tid] = src[t * BB + tid];
    __syncthreads();
    for (int i = tid; i < BB * E_; i += 512) {
      int b = i >> 8, e = i & 255;
      (&sX[0][0])[i] = (_Float16)emb_enc[(long)stok[b] * E_ + e];
    }
    __syncthreads();

    // ---- r,z gates: 4 tiles per wave, fused i+h accumulators ----
    v8f acc0 = {}, acc1 = {}, acc2 = {}, acc3 = {};
    const long jb = (long)(wave << 6);
    #pragma unroll 1
    for (int k = 0; k < E_; k += 32) {
      v16h a = load_a(&sX[0][0], E_, m, hf, k);
      acc0 = wmma16(a, load_b(Wih + (jb +  0) * E_, E_, m, hf, k), acc0);
      acc1 = wmma16(a, load_b(Wih + (jb + 16) * E_, E_, m, hf, k), acc1);
      acc2 = wmma16(a, load_b(Wih + (jb + 32) * E_, E_, m, hf, k), acc2);
      acc3 = wmma16(a, load_b(Wih + (jb + 48) * E_, E_, m, hf, k), acc3);
    }
    #pragma unroll 1
    for (int k = 0; k < H_; k += 32) {
      v16h a = load_a(&sH[0][0], H_, m, hf, k);
      acc0 = wmma16(a, load_b(Whh + (jb +  0) * H_, H_, m, hf, k), acc0);
      acc1 = wmma16(a, load_b(Whh + (jb + 16) * H_, H_, m, hf, k), acc1);
      acc2 = wmma16(a, load_b(Whh + (jb + 32) * H_, H_, m, hf, k), acc2);
      acc3 = wmma16(a, load_b(Whh + (jb + 48) * H_, H_, m, hf, k), acc3);
    }
    {
      v8f accs[4] = {acc0, acc1, acc2, acc3};
      #pragma unroll
      for (int q = 0; q < 4; ++q) {
        int j = jrz + (q << 4);
        #pragma unroll
        for (int r = 0; r < 8; ++r) {
          int row = (hf << 3) + r;
          float g = sigmoidf_(accs[q][r] + bsum_rz[q]);
          if (j < H_) sR[row][j] = (_Float16)g;
          else        sZ[row][j - H_] = (_Float16)g;
        }
      }
    }

    // ---- n gate matmuls: 2 tiles per wave, i and h kept separate ----
    v8f ai0 = {}, ai1 = {}, ah0 = {}, ah1 = {};
    const long jc = (long)(1024 + (wave << 5));
    #pragma unroll 1
    for (int k = 0; k < E_; k += 32) {
      v16h a = load_a(&sX[0][0], E_, m, hf, k);
      ai0 = wmma16(a, load_b(Wih + (jc +  0) * E_, E_, m, hf, k), ai0);
      ai1 = wmma16(a, load_b(Wih + (jc + 16) * E_, E_, m, hf, k), ai1);
    }
    #pragma unroll 1
    for (int k = 0; k < H_; k += 32) {
      v16h a = load_a(&sH[0][0], H_, m, hf, k);
      ah0 = wmma16(a, load_b(Whh + (jc +  0) * H_, H_, m, hf, k), ah0);
      ah1 = wmma16(a, load_b(Whh + (jc + 16) * H_, H_, m, hf, k), ah1);
    }
    __syncthreads();   // sR/sZ ready; all sH reads done

    {
      v8f ais[2] = {ai0, ai1}, ahs[2] = {ah0, ah1};
      #pragma unroll
      for (int q = 0; q < 2; ++q) {
        int jH = jn0 + (q << 4) - 1024;
        #pragma unroll
        for (int r = 0; r < 8; ++r) {
          int row = (hf << 3) + r;
          float rr = (float)sR[row][jH];
          float zz = (float)sZ[row][jH];
          float nn = tanhf_(ais[q][r] + bi_n[q] + rr * (ahs[q][r] + bh_n[q]));
          float ho = (float)sH[row][jH];
          sH[row][jH] = (_Float16)((1.0f - zz) * nn + zz * ho);
        }
      }
    }
    __syncthreads();
  }

  for (int i = tid; i < BB * H_; i += 512) {
    int b = i >> 9, j = i & 511;
    float v = (float)sH[b][j];
    ctx_f32[i] = v;
    ctx_f16[i] = sH[b][j];
    hid_f32[((b << 1) + 0) * H_ + j] = v;
    hid_f32[((b << 1) + 1) * H_ + j] = v;
    hid_f16[((b << 1) + 0) * H_ + j] = sH[b][j];
    hid_f16[((b << 1) + 1) * H_ + j] = sH[b][j];
  }
}

// ---------------- decoder GRU step: 1 block, 16 waves, rows = 32 (B*BEAM) ----------------
__global__ __launch_bounds__(512) void dec_gru_kernel(
    const int* __restrict__ tok, const float* __restrict__ emb_dec,
    const _Float16* __restrict__ ctx_f16,
    const float* __restrict__ hid_f32, const _Float16* __restrict__ hid_f16,
    const _Float16* __restrict__ Wih, const _Float16* __restrict__ Whh,
    const float* __restrict__ bih, const float* __restrict__ bhh,
    _Float16* __restrict__ x_f16, _Float16* __restrict__ f_f16,
    float* __restrict__ newh_f32)
{
  __shared__ __align__(16) _Float16 sR[BB * BEAM_][H_];   // 32 KB
  __shared__ __align__(16) _Float16 sZ[BB * BEAM_][H_];   // 32 KB

  const int tid  = threadIdx.x;
  const int wave = tid >> 5;
  const int lane = tid & 31;
  const int m  = lane & 15;
  const int hf = lane >> 4;
  const int mt = wave & 1;          // M-tile (rows 0..15 / 16..31)
  const int wg = wave >> 1;         // 0..7 column group

  // gather x = [emb | ctx] (f16, ws) and emb/ctx slices of f = [emb | newh | ctx]
  for (int i = tid; i < 32 * EH; i += 512) {
    int r32 = i / EH, c = i - r32 * EH;
    _Float16 v;
    if (c < E_) v = (_Float16)emb_dec[(long)tok[r32] * E_ + c];
    else        v = ctx_f16[(r32 >> 1) * H_ + (c - E_)];
    x_f16[r32 * EH + c] = v;
    f_f16[r32 * F_ + (c < E_ ? c : c + H_)] = v;   // ctx -> offset 768..1279
  }
  __threadfence_block();
  __syncthreads();

  const _Float16* Ax = x_f16   + (long)mt * 16 * EH;
  const _Float16* Ah = hid_f16 + (long)mt * 16 * H_;

  // ---- r,z gates: 8 col-tiles per wave (cols wg*128 .. wg*128+127), fused i+h ----
  {
    v8f acc[8] = {{},{},{},{},{},{},{},{}};
    const long jb = (long)(wg << 7);
    #pragma unroll 1
    for (int k = 0; k < EH; k += 32) {
      v16h a = load_a(Ax, EH, m, hf, k);
      #pragma unroll
      for (int q = 0; q < 8; ++q)
        acc[q] = wmma16(a, load_b(Wih + (jb + (q << 4)) * EH, EH, m, hf, k), acc[q]);
    }
    #pragma unroll 1
    for (int k = 0; k < H_; k += 32) {
      v16h a = load_a(Ah, H_, m, hf, k);
      #pragma unroll
      for (int q = 0; q < 8; ++q)
        acc[q] = wmma16(a, load_b(Whh + (jb + (q << 4)) * H_, H_, m, hf, k), acc[q]);
    }
    #pragma unroll
    for (int q = 0; q < 8; ++q) {
      int j = (wg << 7) + (q << 4) + m;
      float bsum = bih[j] + bhh[j];
      #pragma unroll
      for (int r = 0; r < 8; ++r) {
        int row = mt * 16 + (hf << 3) + r;
        float g = sigmoidf_(acc[q][r] + bsum);
        if (j < H_) sR[row][j] = (_Float16)g;
        else        sZ[row][j - H_] = (_Float16)g;
      }
    }
  }
  __syncthreads();

  // ---- n gate + combine: 4 col-tiles per wave, processed in 2 groups of 2 ----
  #pragma unroll 1
  for (int g = 0; g < 2; ++g) {
    const long jc = (long)(1024 + (wg << 6) + (g << 5));
    v8f ai0 = {}, ai1 = {}, ah0 = {}, ah1 = {};
    #pragma unroll 1
    for (int k = 0; k < EH; k += 32) {
      v16h a = load_a(Ax, EH, m, hf, k);
      ai0 = wmma16(a, load_b(Wih + (jc +  0) * EH, EH, m, hf, k), ai0);
      ai1 = wmma16(a, load_b(Wih + (jc + 16) * EH, EH, m, hf, k), ai1);
    }
    #pragma unroll 1
    for (int k = 0; k < H_; k += 32) {
      v16h a = load_a(Ah, H_, m, hf, k);
      ah0 = wmma16(a, load_b(Whh + (jc +  0) * H_, H_, m, hf, k), ah0);
      ah1 = wmma16(a, load_b(Whh + (jc + 16) * H_, H_, m, hf, k), ah1);
    }
    v8f ais[2] = {ai0, ai1}, ahs[2] = {ah0, ah1};
    #pragma unroll
    for (int q = 0; q < 2; ++q) {
      int j  = (int)jc + (q << 4) + m;
      int jH = j - 1024;
      float bi_ = bih[j], bh_ = bhh[j];
      #pragma unroll
      for (int r = 0; r < 8; ++r) {
        int row = mt * 16 + (hf << 3) + r;
        float rr = (float)sR[row][jH];
        float zz = (float)sZ[row][jH];
        float nn = tanhf_(ais[q][r] + bi_ + rr * (ahs[q][r] + bh_));
        float ho = hid_f32[row * H_ + jH];
        float hn = (1.0f - zz) * nn + zz * ho;
        newh_f32[row * H_ + jH] = hn;
        f_f16[row * F_ + E_ + jH] = (_Float16)hn;
      }
    }
  }
}

// ---------------- logits GEMM: f[32,1280](f16) @ Wout^T + b -> logits[32,V] ----------------
// 500 blocks x 8 waves; each wave owns one 16x16 tile, K=1280 (40 WMMAs).
// Wout f16 (82 MB) is fully L2-resident on MI455X (192 MB L2) across the 31 steps.
__global__ __launch_bounds__(256) void dec_logits_kernel(
    const _Float16* __restrict__ f_f16, const _Float16* __restrict__ Wout,
    const float* __restrict__ b_out, float* __restrict__ logits)
{
  const int tid  = threadIdx.x;
  const int wave = tid >> 5;
  const int lane = tid & 31;
  const int m  = lane & 15;
  const int hf = lane >> 4;
  const int mt = wave & 1;
  const int nbase = blockIdx.x * 64 + (wave >> 1) * 16;

  const _Float16* Abase = f_f16 + (long)mt * 16 * F_;
  const _Float16* Wb    = Wout + (long)nbase * F_;
  v8f acc = {};
  #pragma unroll 2
  for (int k = 0; k < F_; k += 32) {
    __builtin_prefetch(Wb + (long)m * F_ + k + 128, 0, 1);  // stream next weight chunk
    acc = wmma16(load_a(Abase, F_, m, hf, k), load_b(Wb, F_, m, hf, k), acc);
  }
  int col = nbase + m;
  float bo = b_out[col];
  #pragma unroll
  for (int r = 0; r < 8; ++r) {
    int row = mt * 16 + (hf << 3) + r;
    logits[(long)row * V_ + col] = acc[r] + bo;
  }
}

// ---------------- per-row softmax stats: max & 1/sum(exp) ----------------
__global__ __launch_bounds__(256) void dec_stats_kernel(
    const float* __restrict__ logits, float* __restrict__ rowmax, float* __restrict__ rowinv)
{
  __shared__ float red[256];
  const int row = blockIdx.x;
  const float* p = logits + (long)row * V_;
  const int tid = threadIdx.x;
  float mx = -3.4e38f;
  for (int i = tid; i < V_; i += 256) mx = fmaxf(mx, p[i]);
  red[tid] = mx; __syncthreads();
  for (int s = 128; s > 0; s >>= 1) { if (tid < s) red[tid] = fmaxf(red[tid], red[tid + s]); __syncthreads(); }
  mx = red[0]; __syncthreads();
  float sm = 0.0f;
  for (int i = tid; i < V_; i += 256) sm += __expf(p[i] - mx);
  red[tid] = sm; __syncthreads();
  for (int s = 128; s > 0; s >>= 1) { if (tid < s) red[tid] += red[tid + s]; __syncthreads(); }
  if (tid == 0) { rowmax[row] = mx; rowinv[row] = __builtin_amdgcn_rcpf(red[0]); }
}

// ---------------- top-2 of BEAM*V + beam bookkeeping (one block per batch row) -----------
struct Top2 { float v0, v1; int i0, i1; };
__device__ __forceinline__ Top2 top2_merge(Top2 a, Top2 b) {
  Top2 o;
  if (a.v0 >= b.v0) {
    o.v0 = a.v0; o.i0 = a.i0;
    bool t = a.v1 >= b.v0; o.v1 = t ? a.v1 : b.v0; o.i1 = t ? a.i1 : b.i0;
  } else {
    o.v0 = b.v0; o.i0 = b.i0;
    bool t = b.v1 >= a.v0; o.v1 = t ? b.v1 : a.v0; o.i1 = t ? b.i1 : a.i0;
  }
  return o;
}

__global__ __launch_bounds__(256) void dec_topk_kernel(
    const float* __restrict__ logits, const float* __restrict__ rowmax,
    const float* __restrict__ rowinv, const float* __restrict__ newh,
    int* __restrict__ tok, float* __restrict__ prob, int* __restrict__ fin,
    int* __restrict__ th, float* __restrict__ sh,
    float* __restrict__ hid_f32, _Float16* __restrict__ hid_f16, int t)
{
  __shared__ Top2 red[256];
  __shared__ int   th_old[BEAM_][MAXLEN];
  __shared__ float sh_old[BEAM_][MAXLEN];
  __shared__ int   bc_parent[BEAM_], bc_token[BEAM_], bc_finp[BEAM_];
  __shared__ float bc_topv[BEAM_];

  const int b = blockIdx.x;
  const int tid = threadIdx.x;
  const float pr0 = prob[b * 2 + 0], pr1 = prob[b * 2 + 1];
  const int   f0  = fin[b * 2 + 0],  f1  = fin[b * 2 + 1];
  const float mx0 = rowmax[b * 2 + 0], mx1 = rowmax[b * 2 + 1];
  const float iv0 = rowinv[b * 2 + 0], iv1 = rowinv[b * 2 + 1];

  Top2 best; best.v0 = -3.4e38f; best.v1 = -3.4e38f; best.i0 = 0; best.i1 = 0;
  for (int idx = tid; idx < 2 * V_; idx += 256) {
    int k = (idx >= V_) ? 1 : 0;
    int v = idx - k * V_;
    float pr = k ? pr1 : pr0;
    int   fn = k ? f1  : f0;
    float c;
    if (fn) c = (v == EOS_) ? pr : NEG_;
    else {
      float l = logits[(long)(b * 2 + k) * V_ + v];
      c = __expf(l - (k ? mx1 : mx0)) * (k ? iv1 : iv0) * pr;
    }
    Top2 e; e.v0 = c; e.i0 = idx; e.v1 = -3.4e38f; e.i1 = 0;
    best = top2_merge(best, e);
  }
  red[tid] = best; __syncthreads();
  for (int s = 128; s > 0; s >>= 1) {
    if (tid < s) red[tid] = top2_merge(red[tid], red[tid + s]);
    __syncthreads();
  }

  if (tid < BEAM_ * MAXLEN) {                 // snapshot histories before overwrite
    int k = tid / MAXLEN, i = tid % MAXLEN;
    th_old[k][i] = th[(b * BEAM_ + k) * MAXLEN + i];
    sh_old[k][i] = sh[(b * BEAM_ + k) * MAXLEN + i];
  }
  if (tid == 0) {
    int   topi[2] = { red[0].i0, red[0].i1 };
    float topv[2] = { red[0].v0, red[0].v1 };
    for (int k = 0; k < 2; ++k) {
      int par = (topi[k] >= V_) ? 1 : 0;
      int tkn = topi[k] - par * V_;
      int fp  = par ? f1 : f0;
      bc_parent[k] = par; bc_token[k] = tkn; bc_finp[k] = fp; bc_topv[k] = topv[k];
      tok[b * 2 + k]  = tkn;
      prob[b * 2 + k] = topv[k];
      fin[b * 2 + k]  = fp | (tkn == EOS_);
    }
  }
  __syncthreads();

  if (tid < BEAM_ * MAXLEN) {                 // gathered histories + position-t update
    int k = tid / MAXLEN, i = tid % MAXLEN;
    int par = bc_parent[k];
    int   tv = th_old[par][i];
    float sv = sh_old[par][i];
    if (i == t) {
      if (!bc_finp[k]) tv = bc_token[k];
      sv = bc_finp[k] ? 0.0f : bc_topv[k];
    }
    th[(b * BEAM_ + k) * MAXLEN + i] = tv;
    sh[(b * BEAM_ + k) * MAXLEN + i] = sv;
  }
  for (int i = tid; i < BEAM_ * H_; i += 256) {   // hid <- newh[parent]
    int k = i >> 9, j = i & 511;
    float v = newh[(b * 2 + bc_parent[k]) * H_ + j];
    hid_f32[(b * 2 + k) * H_ + j] = v;
    hid_f16[(b * 2 + k) * H_ + j] = (_Float16)v;
  }
}

// ---------------- finalize: pick best beam, emit tokens then scores ----------------
__global__ void finalize_kernel(const float* __restrict__ prob, const int* __restrict__ th,
                                const float* __restrict__ sh, float* __restrict__ out)
{
  int i = blockIdx.x * blockDim.x + threadIdx.x;
  if (i >= BB * MAXLEN) return;
  int b = i / MAXLEN, j = i % MAXLEN;
  int best = (prob[b * 2 + 1] > prob[b * 2 + 0]) ? 1 : 0;
  out[i]               = (float)th[(b * BEAM_ + best) * MAXLEN + j];
  out[BB * MAXLEN + i] = sh[(b * BEAM_ + best) * MAXLEN + j];
}

// =====================================================================================
extern "C" void kernel_launch(void* const* d_in, const int* in_sizes, int n_in,
                              void* d_out, int out_size, void* d_ws, size_t ws_size,
                              hipStream_t stream)
{
  (void)in_sizes; (void)n_in; (void)out_size; (void)ws_size;

  const int*   src     = (const int*)  d_in[0];
  const float* emb_enc = (const float*)d_in[1];
  const float* enc_Wih = (const float*)d_in[2];
  const float* enc_Whh = (const float*)d_in[3];
  const float* enc_bih = (const float*)d_in[4];
  const float* enc_bhh = (const float*)d_in[5];
  const float* emb_dec = (const float*)d_in[6];
  const float* dec_Wih = (const float*)d_in[7];
  const float* dec_Whh = (const float*)d_in[8];
  const float* dec_bih = (const float*)d_in[9];
  const float* dec_bhh = (const float*)d_in[10];
  const float* W_out   = (const float*)d_in[11];
  const float* b_out   = (const float*)d_in[12];

  char* ws = (char*)d_ws;
  size_t off = 0;
  auto carve = [&](size_t bytes) -> void* {
    void* p = ws + off;
    off = (off + bytes + 255) & ~(size_t)255;
    return p;
  };
  _Float16* Wout16 = (_Float16*)carve((size_t)V_ * F_ * 2);   // 81.92 MB (L2-resident)
  _Float16* dWih16 = (_Float16*)carve((size_t)G3 * EH * 2);
  _Float16* dWhh16 = (_Float16*)carve((size_t)G3 * H_ * 2);
  _Float16* eWih16 = (_Float16*)carve((size_t)G3 * E_ * 2);
  _Float16* eWhh16 = (_Float16*)carve((size_t)G3 * H_ * 2);
  float*    ctx32  = (float*)   carve((size_t)BB * H_ * 4);
  _Float16* ctx16  = (_Float16*)carve((size_t)BB * H_ * 2);
  float*    hid32  = (float*)   carve((size_t)BB * BEAM_ * H_ * 4);
  _Float16* hid16  = (_Float16*)carve((size_t)BB * BEAM_ * H_ * 2);
  float*    newh32 = (float*)   carve((size_t)BB * BEAM_ * H_ * 4);
  _Float16* x16    = (_Float16*)carve((size_t)BB * BEAM_ * EH * 2);
  _Float16* fbuf16 = (_Float16*)carve((size_t)BB * BEAM_ * F_ * 2);
  float*    logits = (float*)   carve((size_t)BB * BEAM_ * V_ * 4); // 4.1 MB
  float*    rowmax = (float*)   carve(32 * 4);
  float*    rowinv = (float*)   carve(32 * 4);
  int*      tok    = (int*)     carve(32 * 4);
  float*    prob   = (float*)   carve(32 * 4);
  int*      fin    = (int*)     carve(32 * 4);
  int*      th     = (int*)     carve((size_t)BB * BEAM_ * MAXLEN * 4);
  float*    sh     = (float*)   carve((size_t)BB * BEAM_ * MAXLEN * 4);

  // 1) weights -> f16 (W_out f16 fits in MI455X's 192 MB L2 -> 31 GEMMs run out of L2)
  f32_to_f16_kernel<<<2048, 256, 0, stream>>>(W_out,   Wout16, (long)V_ * F_);
  f32_to_f16_kernel<<<256,  256, 0, stream>>>(dec_Wih, dWih16, (long)G3 * EH);
  f32_to_f16_kernel<<<256,  256, 0, stream>>>(dec_Whh, dWhh16, (long)G3 * H_);
  f32_to_f16_kernel<<<256,  256, 0, stream>>>(enc_Wih, eWih16, (long)G3 * E_);
  f32_to_f16_kernel<<<256,  256, 0, stream>>>(enc_Whh, eWhh16, (long)G3 * H_);

  // 2) beam state init
  init_state_kernel<<<4, 256, 0, stream>>>(tok, prob, fin, th, sh);

  // 3) encoder: persistent single-workgroup GRU over 128 steps (latency-bound chain)
  enc_gru_kernel<<<1, 512, 0, stream>>>(src, emb_enc, eWih16, eWhh16, enc_bih, enc_bhh,
                                        ctx32, ctx16, hid32, hid16);

  // 4) 31 decoder steps
  for (int t = 1; t < MAXLEN; ++t) {
    dec_gru_kernel<<<1, 512, 0, stream>>>(tok, emb_dec, ctx16, hid32, hid16,
                                          dWih16, dWhh16, dec_bih, dec_bhh,
                                          x16, fbuf16, newh32);
    dec_logits_kernel<<<V_ / 64, 256, 0, stream>>>(fbuf16, Wout16, b_out, logits);
    dec_stats_kernel<<<32, 256, 0, stream>>>(logits, rowmax, rowinv);
    dec_topk_kernel<<<BB, 256, 0, stream>>>(logits, rowmax, rowinv, newh32,
                                            tok, prob, fin, th, sh, hid32, hid16, t);
  }

  // 5) finalize: tokens (as float) then scores
  finalize_kernel<<<2, 256, 0, stream>>>(prob, th, sh, (float*)d_out);
}
